// GraphUpdateBlock_51376398794962
// MI455X (gfx1250) — compile-verified, matrix operand order checked by reading, not run on hardware
//
#include <hip/hip_runtime.h>
#include <cstdint>

// ---------------- problem constants ----------------
constexpr int BB   = 2;
constexpr int NN   = 8192;
constexpr int EG   = 65536;
constexpr int Bn   = BB * NN;          // 16384
constexpr int DD   = 256;
constexpr int HH   = 4;
constexpr int CC   = 64;
constexpr int EBASE = BB * EG;         // 131072
constexpr int ETOT  = EBASE + Bn;      // 147456
constexpr int KPAD1 = 288;             // 258 -> 288 (9 * 32)

typedef __attribute__((ext_vector_type(16))) __bf16 v16bf;
typedef __attribute__((ext_vector_type(8)))  float  v8f;

union AFrag { uint4 q[2]; v16bf v; };

__device__ __forceinline__ unsigned short f2bf(float f) {
  union { float f; unsigned u; } x; x.f = f;
  unsigned u = x.u;
  unsigned r = u + 0x7FFFu + ((u >> 16) & 1u);   // RNE
  return (unsigned short)(r >> 16);
}
__device__ __forceinline__ unsigned fenc(float f) {   // order-preserving float->uint
  union { float f; unsigned u; } x; x.f = f;
  return (x.u & 0x80000000u) ? ~x.u : (x.u | 0x80000000u);
}
__device__ __forceinline__ float fdec(unsigned k) {
  union { float f; unsigned u; } x;
  x.u = (k & 0x80000000u) ? (k & 0x7FFFFFFFu) : ~k;
  return x.f;
}
__device__ __forceinline__ float sigf(float x) { return 1.0f / (1.0f + __expf(-x)); }

// ---------------- utility kernels ----------------
__global__ void zero32(unsigned* p, int n) {
  int t = blockIdx.x * blockDim.x + threadIdx.x;
  if (t < n) p[t] = 0u;
}

__global__ void f32_to_bf16_k(const float* __restrict__ in, unsigned short* __restrict__ out, int n) {
  int t = blockIdx.x * blockDim.x + threadIdx.x;
  if (t < n) out[t] = f2bf(in[t]);
}

// x = concat(e_proj[Bn,2], f_Lt[Bn,256]) zero-padded to [Bn,288] bf16
__global__ void build_xpad_k(const float* __restrict__ ep, const float* __restrict__ fl,
                             unsigned short* __restrict__ out) {
  int t = blockIdx.x * blockDim.x + threadIdx.x;
  if (t >= Bn * KPAD1) return;
  int row = t / KPAD1, j = t - row * KPAD1;
  float v = 0.0f;
  if (j < 2)        v = ep[row * 2 + j];
  else if (j < 258) v = fl[row * 256 + (j - 2)];
  out[t] = f2bf(v);
}

// pack f32 W[Ksrc,N] -> bf16 fragment-major layout, K zero-padded to Kpad
__global__ void pack_weight_k(const float* __restrict__ W, unsigned short* __restrict__ out,
                              int Ksrc, int Kpad, int N) {
  int KB = Kpad >> 5, NB = N >> 4;
  int t = blockIdx.x * blockDim.x + threadIdx.x;
  if (t >= NB * KB * 32) return;
  int lane = t & 31;
  int kblk = (t >> 5) % KB;
  int nblk = t / (32 * KB);
  int hf = lane >> 4;
  int n = nblk * 16 + (lane & 15);
  unsigned short* o = out + ((size_t)(nblk * KB + kblk) * 32 + lane) * 16;
#pragma unroll
  for (int j = 0; j < 16; ++j) {
    int v = j >> 1, p = j & 1;
    int k = kblk * 32 + v * 2 + p + ((v >= 4) ? 8 : 0) + hf * 8;
    float f = (k < Ksrc) ? W[(size_t)k * N + n] : 0.0f;
    o[j] = f2bf(f);
  }
}

// ---------------- WMMA GEMM: C[M,N] = A[M,KB*32](bf16) @ Bpack + bias, act ----------------
// 8 waves: 4 (M) x 2 (N); wave tile 16x64 (4 WMMA accumulators). K-loop trip
// count is a template constant and fully unrolled -> straight-line loads+WMMA,
// letting the scheduler hoist next-step global_load_b128 above current WMMAs.
template <int KB>
__global__ __launch_bounds__(256) void gemm_bf16_k(const unsigned short* __restrict__ A,
                                                   const unsigned short* __restrict__ Bp,
                                                   float* __restrict__ C,
                                                   const float* __restrict__ bias,
                                                   int N, int act) {
  constexpr int Kpad = KB * 32;
  int lane = threadIdx.x & 31;
  int wave = threadIdx.x >> 5;
  int rowbase = blockIdx.x * 64 + (wave >> 1) * 16;
  int colbase = blockIdx.y * 128 + (wave & 1) * 64;
  int hf = lane >> 4;
  int arow = rowbase + (lane & 15);
  const char* abase = (const char*)(A + (size_t)arow * Kpad);
  const char* bbase = (const char*)Bp + ((size_t)(colbase >> 4) * KB + 0) * 1024 + (size_t)lane * 32;

  v8f acc[4] = {};
#pragma unroll
  for (int kb = 0; kb < KB; ++kb) {
    AFrag a;
    const char* ap = abase + kb * 64;
    a.q[0] = *(const uint4*)(ap + hf * 16);
    a.q[1] = *(const uint4*)(ap + 32 + hf * 16);
    AFrag b[4];
#pragma unroll
    for (int nf = 0; nf < 4; ++nf) {
      const char* bp = bbase + ((size_t)nf * KB + kb) * 1024;
      b[nf].q[0] = *(const uint4*)bp;
      b[nf].q[1] = *(const uint4*)(bp + 16);
    }
#pragma unroll
    for (int nf = 0; nf < 4; ++nf) {
      acc[nf] = __builtin_amdgcn_wmma_f32_16x16x32_bf16(
          false, a.v, false, b[nf].v, (short)0, acc[nf], false, false);
    }
  }

#pragma unroll
  for (int nf = 0; nf < 4; ++nf) {
#pragma unroll
    for (int v = 0; v < 8; ++v) {
      int r = rowbase + v + hf * 8;
      int c = colbase + nf * 16 + (lane & 15);
      float val = acc[nf][v];
      if (bias) val += bias[c];
      if (act == 1) val = val * sigf(val);   // SiLU
      C[(size_t)r * N + c] = val;
    }
  }
}

// ---------------- LayerNorm (+optional pre-bias, SiLU, residual); wave per row ----------------
__global__ __launch_bounds__(256) void ln_k(const float* __restrict__ in,
                                            const float* __restrict__ prebias,
                                            const float* __restrict__ g,
                                            const float* __restrict__ b,
                                            const float* __restrict__ resid,
                                            float* __restrict__ out,
                                            int do_silu) {
  int lane = threadIdx.x & 31;
  int row = blockIdx.x * 8 + (threadIdx.x >> 5);
  const float* rp = in + (size_t)row * 256;
  float xv[8];
  float s = 0.0f;
#pragma unroll
  for (int j = 0; j < 8; ++j) {
    int c = lane + j * 32;
    float v = rp[c];
    if (prebias) v += prebias[c];
    xv[j] = v; s += v;
  }
#pragma unroll
  for (int m = 16; m >= 1; m >>= 1) s += __shfl_xor(s, m, 32);
  float mu = s * (1.0f / 256.0f);
  float vs = 0.0f;
#pragma unroll
  for (int j = 0; j < 8; ++j) { float d = xv[j] - mu; vs += d * d; }
#pragma unroll
  for (int m = 16; m >= 1; m >>= 1) vs += __shfl_xor(vs, m, 32);
  float rs = rsqrtf(vs * (1.0f / 256.0f) + 1e-5f);
#pragma unroll
  for (int j = 0; j < 8; ++j) {
    int c = lane + j * 32;
    float y = (xv[j] - mu) * rs * g[c] + b[c];
    if (do_silu) y = y * sigf(y);
    if (resid) y += resid[(size_t)row * 256 + c];
    out[(size_t)row * 256 + c] = y;
  }
}

// ---------------- edge decoding ----------------
__device__ __forceinline__ void edge_sd(int e, const int* __restrict__ edges, int& src, int& dst) {
  if (e < EBASE) {
    int b = e >> 16, j = e & 65535;
    src = edges[(size_t)b * 2 * EG + j] + b * NN;
    dst = edges[(size_t)b * 2 * EG + EG + j] + b * NN;
  } else {
    src = dst = e - EBASE;
  }
}

__global__ void degfill_k(const int* __restrict__ edges, const float* __restrict__ ea,
                          float* __restrict__ deg, float* __restrict__ fill) {
  int e = blockIdx.x * blockDim.x + threadIdx.x;
  if (e >= EBASE) return;
  int src, dst; edge_sd(e, edges, src, dst);
  atomicAdd(&deg[dst], 1.0f);
  atomicAdd(&fill[dst * 3 + 0], ea[(size_t)e * 3 + 0]);
  atomicAdd(&fill[dst * 3 + 1], ea[(size_t)e * 3 + 1]);
  atomicAdd(&fill[dst * 3 + 2], ea[(size_t)e * 3 + 2]);
}

__global__ void filldiv_k(const float* __restrict__ deg, float* __restrict__ fill) {
  int i = blockIdx.x * blockDim.x + threadIdx.x;
  if (i >= Bn) return;
  float d = fmaxf(deg[i], 1.0f);
  fill[i * 3 + 0] /= d; fill[i * 3 + 1] /= d; fill[i * 3 + 2] /= d;
}

// logits + segment max; one thread per (edge, head). We/att staged in LDS.
__global__ __launch_bounds__(256) void edge_logits_k(const float* __restrict__ XL,
                                                     const float* __restrict__ XR,
                                                     const float* __restrict__ We,
                                                     const float* __restrict__ att,
                                                     const float* __restrict__ fill,
                                                     const int* __restrict__ edges,
                                                     const float* __restrict__ eattr,
                                                     float* __restrict__ logits,
                                                     unsigned* __restrict__ maxb) {
  __shared__ float sWe[768];
  __shared__ float sAtt[256];
  int t = threadIdx.x;
  sWe[t] = We[t]; sWe[256 + t] = We[256 + t]; sWe[512 + t] = We[512 + t];
  sAtt[t] = att[t];
  __syncthreads();
  int tid = blockIdx.x * 256 + t;
  if (tid >= ETOT * HH) return;
  int h = tid & 3, e = tid >> 2;
  int src, dst; edge_sd(e, edges, src, dst);
  float ea0, ea1, ea2;
  if (e < EBASE) { const float* p = eattr + (size_t)e * 3; ea0 = p[0]; ea1 = p[1]; ea2 = p[2]; }
  else           { const float* p = fill + (size_t)(e - EBASE) * 3; ea0 = p[0]; ea1 = p[1]; ea2 = p[2]; }
  const float* xlp = XL + (size_t)src * 256 + h * 64;
  const float* xrp = XR + (size_t)dst * 256 + h * 64;
  float lg = 0.0f;
#pragma unroll 8
  for (int c = 0; c < 64; ++c) {
    int idx = h * 64 + c;
    float eap = ea0 * sWe[idx] + ea1 * sWe[256 + idx] + ea2 * sWe[512 + idx];
    float m = xlp[c] + xrp[c] + eap;
    m = (m > 0.0f) ? m : 0.2f * m;         // leaky_relu 0.2
    lg += m * sAtt[idx];
  }
  logits[tid] = lg;
  atomicMax(&maxb[(size_t)dst * 4 + h], fenc(lg));
}

__global__ void edge_exp_k(const int* __restrict__ edges, float* __restrict__ logits,
                           const unsigned* __restrict__ maxb, float* __restrict__ den) {
  int tid = blockIdx.x * blockDim.x + threadIdx.x;
  if (tid >= ETOT * HH) return;
  int h = tid & 3, e = tid >> 2;
  int src, dst; edge_sd(e, edges, src, dst);
  float ex = __expf(logits[tid] - fdec(maxb[(size_t)dst * 4 + h]));
  logits[tid] = ex;
  atomicAdd(&den[(size_t)dst * 4 + h], ex);
}

__global__ void edge_agg_k(const int* __restrict__ edges, const float* __restrict__ logits,
                           const float* __restrict__ den, const float* __restrict__ XL,
                           float* __restrict__ out) {
  int tid = blockIdx.x * blockDim.x + threadIdx.x;
  if (tid >= ETOT * HH) return;
  int h = tid & 3, e = tid >> 2;
  int src, dst; edge_sd(e, edges, src, dst);
  float alpha = logits[tid] / (den[(size_t)dst * 4 + h] + 1e-16f);
  const float* xlp = XL + (size_t)src * 256 + h * 64;
  float* op = out + (size_t)dst * 256 + h * 64;
#pragma unroll 8
  for (int c = 0; c < 64; ++c) atomicAdd(&op[c], alpha * xlp[c]);
}

// ---------------- GRU gates ----------------
__global__ void gru_k(const float* __restrict__ gi, const float* __restrict__ gh,
                      const float* __restrict__ hp, float* __restrict__ out) {
  int t = blockIdx.x * blockDim.x + threadIdx.x;
  if (t >= Bn * 256) return;
  int row = t >> 8, c = t & 255;
  size_t b = (size_t)row * 768 + c;
  float r = sigf(gi[b] + gh[b]);
  float z = sigf(gi[b + 256] + gh[b + 256]);
  float n = tanhf(gi[b + 512] + r * gh[b + 512]);
  out[t] = (1.0f - z) * n + z * hp[t];
}

// ---------------- tiny head GEMVs ----------------
__global__ void head_w_k(const float* __restrict__ wz, const float* __restrict__ W2,
                         const float* __restrict__ b2, float* __restrict__ w_out) {
  int t = blockIdx.x * blockDim.x + threadIdx.x;
  if (t >= Bn) return;
  const float* rp = wz + (size_t)t * 256;
  float s = b2[0];
  for (int c = 0; c < 256; ++c) s += rp[c] * W2[c * 2];
  w_out[t] = sigf(s) * 0.95f + 0.05f;
}

__global__ void head_res_k(const float* __restrict__ rz, const float* __restrict__ W2,
                           const float* __restrict__ b2, float* __restrict__ res_out) {
  int t = blockIdx.x * blockDim.x + threadIdx.x;
  if (t >= Bn) return;
  const float* rp = rz + (size_t)t * 256;
  float s0 = b2[0], s1 = b2[1];
  for (int c = 0; c < 256; ++c) { float v = rp[c]; s0 += v * W2[c * 2]; s1 += v * W2[c * 2 + 1]; }
  res_out[t * 2] = s0; res_out[t * 2 + 1] = s1;
}

__global__ void head_ap_k(const float* __restrict__ z, const float* __restrict__ W2,
                          const float* __restrict__ b2, float* __restrict__ acc) {
  int t = blockIdx.x * blockDim.x + threadIdx.x;
  if (t >= Bn) return;
  const float* rp = z + (size_t)t * 128;
  float s = b2[0];
  for (int c = 0; c < 128; ++c) s += rp[c] * W2[c];
  atomicAdd(&acc[t >> 13], sigf(s));
}

__global__ void ap_final_k(const float* __restrict__ acc, float* __restrict__ out) {
  int t = threadIdx.x;
  if (t < BB) out[t] = acc[t] * (0.1f / (float)NN) + 0.0001f;
}

__global__ void head_ad_k(const float* __restrict__ z, const float* __restrict__ W2,
                          const float* __restrict__ b2, float* __restrict__ ad_out) {
  int t = blockIdx.x * blockDim.x + threadIdx.x;
  if (t >= Bn) return;
  const float* rp = z + (size_t)t * 128;
  float s = b2[0];
  for (int c = 0; c < 128; ++c) s += rp[c] * W2[c];
  ad_out[t] = sigf(s) * 0.1f + 0.0001f;
}

// ---------------- host orchestration ----------------
static inline int gdiv(int a, int b) { return (a + b - 1) / b; }

extern "C" void kernel_launch(void* const* d_in, const int* in_sizes, int n_in,
                              void* d_out, int out_size, void* d_ws, size_t ws_size,
                              hipStream_t stream) {
  (void)in_sizes; (void)n_in; (void)out_size; (void)ws_size;
  const float* h_in    = (const float*)d_in[0];
  const float* e_proj  = (const float*)d_in[1];
  const float* f_Lt    = (const float*)d_in[2];
  const int*   edges   = (const int*)d_in[3];
  const float* eattr   = (const float*)d_in[4];
#define P(i) ((const float*)d_in[i])
  // param indices (dict order, offset by 5)
  const float *c1_Wl=P(5), *c1_Wr=P(6), *c1_We=P(7), *c1_att=P(8), *c1_b=P(9);
  const float *n1_g=P(10), *n1_b=P(11);
  const float *c2_Wl=P(12), *c2_Wr=P(13), *c2_We=P(14), *c2_att=P(15), *c2_b=P(16);
  const float *n2_g=P(17), *n2_b=P(18);
  const float *proj_W=P(19), *proj_b=P(20), *ng_g=P(21), *ng_b=P(22);
  const float *gru_Wih=P(23), *gru_Whh=P(24), *gru_bih=P(25), *gru_bhh=P(26);
  const float *nh_g=P(27), *nh_b=P(28);
  const float *wh_W1=P(29), *wh_b1=P(30), *wh_g=P(31), *wh_bn=P(32), *wh_W2=P(33), *wh_b2=P(34);
  const float *rh_W1=P(35), *rh_b1=P(36), *rh_W2=P(37), *rh_b2=P(38);
  const float *ap_W1=P(39), *ap_b1=P(40), *ap_W2=P(41), *ap_b2=P(42);
  const float *ad_W1=P(43), *ad_b1=P(44), *ad_W2=P(45), *ad_b2=P(46);
#undef P

  // workspace carve-out (256B aligned)
  size_t off = 0;
  auto alloc = [&](size_t bytes) -> char* {
    char* p = (char*)d_ws + off;
    off = (off + bytes + 255) & ~(size_t)255;
    return p;
  };
  unsigned short* XPAD = (unsigned short*)alloc((size_t)Bn * KPAD1 * 2);
  unsigned short* ABF0 = (unsigned short*)alloc((size_t)Bn * 256 * 2);
  unsigned short* ABF1 = (unsigned short*)alloc((size_t)Bn * 256 * 2);
  unsigned short* Wl1p = (unsigned short*)alloc((size_t)KPAD1 * 256 * 2);
  unsigned short* Wr1p = (unsigned short*)alloc((size_t)KPAD1 * 256 * 2);
  unsigned short* Wl2p = (unsigned short*)alloc((size_t)256 * 256 * 2);
  unsigned short* Wr2p = (unsigned short*)alloc((size_t)256 * 256 * 2);
  unsigned short* PROJP= (unsigned short*)alloc((size_t)256 * 256 * 2);
  unsigned short* WHP  = (unsigned short*)alloc((size_t)256 * 256 * 2);
  unsigned short* RHP  = (unsigned short*)alloc((size_t)256 * 256 * 2);
  unsigned short* WIHP = (unsigned short*)alloc((size_t)256 * 768 * 2);
  unsigned short* WHHP = (unsigned short*)alloc((size_t)256 * 768 * 2);
  unsigned short* APP  = (unsigned short*)alloc((size_t)256 * 128 * 2);
  unsigned short* ADP  = (unsigned short*)alloc((size_t)256 * 128 * 2);
  float* XL   = (float*)alloc((size_t)Bn * 256 * 4);
  float* XR   = (float*)alloc((size_t)Bn * 256 * 4);
  float* GOUT = (float*)alloc((size_t)Bn * 256 * 4);
  float* Y1   = (float*)alloc((size_t)Bn * 256 * 4);
  float* T0   = (float*)alloc((size_t)Bn * 256 * 4);
  float* GI   = (float*)alloc((size_t)Bn * 768 * 4);
  float* GH   = (float*)alloc((size_t)Bn * 768 * 4);
  float* LOGB = (float*)alloc((size_t)ETOT * HH * 4);
  unsigned* MAXB = (unsigned*)alloc((size_t)Bn * HH * 4);
  float* DEN  = (float*)alloc((size_t)Bn * HH * 4);
  float* DEG  = (float*)alloc((size_t)Bn * 4);
  float* FILL = (float*)alloc((size_t)Bn * 3 * 4);
  float* H128 = (float*)alloc((size_t)Bn * 128 * 4);
  float* APACC= (float*)alloc(256);

  float* o_h   = (float*)d_out;
  float* o_res = o_h + (size_t)Bn * 256;
  float* o_w   = o_res + (size_t)Bn * 2;
  float* o_ap  = o_w + Bn;
  float* o_ad  = o_ap + BB;

  const int TPB = 256;
  auto Z = [&](void* p, int n) { zero32<<<gdiv(n, TPB), TPB, 0, stream>>>((unsigned*)p, n); };

  // --- weight packing (bf16, fragment-major) ---
  auto PACK = [&](const float* W, unsigned short* o, int Ks, int Kp, int N) {
    int tot = (N / 16) * (Kp / 32) * 32;
    pack_weight_k<<<gdiv(tot, TPB), TPB, 0, stream>>>(W, o, Ks, Kp, N);
  };
  PACK(c1_Wl, Wl1p, 258, KPAD1, 256);  PACK(c1_Wr, Wr1p, 258, KPAD1, 256);
  PACK(c2_Wl, Wl2p, 256, 256, 256);    PACK(c2_Wr, Wr2p, 256, 256, 256);
  PACK(proj_W, PROJP, 256, 256, 256);
  PACK(gru_Wih, WIHP, 256, 256, 768);  PACK(gru_Whh, WHHP, 256, 256, 768);
  PACK(wh_W1, WHP, 256, 256, 256);     PACK(rh_W1, RHP, 256, 256, 256);
  PACK(ap_W1, APP, 256, 256, 128);     PACK(ad_W1, ADP, 256, 256, 128);

  build_xpad_k<<<gdiv(Bn * KPAD1, TPB), TPB, 0, stream>>>(e_proj, f_Lt, XPAD);

  auto GEMM = [&](const unsigned short* A, const unsigned short* Bp, float* C,
                  const float* bias, int Kpad, int N, int act) {
    dim3 grid(Bn / 64, N / 128);
    if (Kpad == KPAD1)
      gemm_bf16_k<9><<<grid, 256, 0, stream>>>(A, Bp, C, bias, N, act);
    else
      gemm_bf16_k<8><<<grid, 256, 0, stream>>>(A, Bp, C, bias, N, act);
  };
  auto CVT = [&](const float* in, unsigned short* o) {
    f32_to_bf16_k<<<gdiv(Bn * 256, TPB), TPB, 0, stream>>>(in, o, Bn * 256);
  };
  const int EHT = ETOT * HH;

  // --- self-loop fill (mean edge attr per dst) ---
  Z(DEG, Bn); Z(FILL, Bn * 3);
  degfill_k<<<gdiv(EBASE, TPB), TPB, 0, stream>>>(edges, eattr, DEG, FILL);
  filldiv_k<<<gdiv(Bn, TPB), TPB, 0, stream>>>(DEG, FILL);

  // --- GATv2 layer 1 ---
  GEMM(XPAD, Wl1p, XL, nullptr, KPAD1, 256, 0);
  GEMM(XPAD, Wr1p, XR, nullptr, KPAD1, 256, 0);
  Z(MAXB, Bn * HH); Z(DEN, Bn * HH); Z(GOUT, Bn * 256);
  edge_logits_k<<<gdiv(EHT, TPB), TPB, 0, stream>>>(XL, XR, c1_We, c1_att, FILL, edges, eattr, LOGB, MAXB);
  edge_exp_k<<<gdiv(EHT, TPB), TPB, 0, stream>>>(edges, LOGB, MAXB, DEN);
  edge_agg_k<<<gdiv(EHT, TPB), TPB, 0, stream>>>(edges, LOGB, DEN, XL, GOUT);
  ln_k<<<Bn / 8, 256, 0, stream>>>(GOUT, c1_b, n1_g, n1_b, nullptr, Y1, 1);   // y1 = silu(ln(gat1+b))

  // --- GATv2 layer 2 ---
  CVT(Y1, ABF0);
  GEMM(ABF0, Wl2p, XL, nullptr, 256, 256, 0);
  GEMM(ABF0, Wr2p, XR, nullptr, 256, 256, 0);
  Z(MAXB, Bn * HH); Z(DEN, Bn * HH); Z(GOUT, Bn * 256);
  edge_logits_k<<<gdiv(EHT, TPB), TPB, 0, stream>>>(XL, XR, c2_We, c2_att, FILL, edges, eattr, LOGB, MAXB);
  edge_exp_k<<<gdiv(EHT, TPB), TPB, 0, stream>>>(edges, LOGB, MAXB, DEN);
  edge_agg_k<<<gdiv(EHT, TPB), TPB, 0, stream>>>(edges, LOGB, DEN, XL, GOUT);
  ln_k<<<Bn / 8, 256, 0, stream>>>(GOUT, c2_b, n2_g, n2_b, Y1, T0, 1);        // y2 = silu(ln(..)) + y1

  // --- proj + group LN ---
  CVT(T0, ABF0);
  GEMM(ABF0, PROJP, GOUT, proj_b, 256, 256, 0);
  ln_k<<<Bn / 8, 256, 0, stream>>>(GOUT, nullptr, ng_g, ng_b, nullptr, T0, 0); // xs

  // --- GRU ---
  CVT(T0, ABF0);
  CVT(h_in, ABF1);
  GEMM(ABF0, WIHP, GI, gru_bih, 256, 768, 0);
  GEMM(ABF1, WHHP, GH, gru_bhh, 256, 768, 0);
  gru_k<<<gdiv(Bn * 256, TPB), TPB, 0, stream>>>(GI, GH, h_in, XL);
  ln_k<<<Bn / 8, 256, 0, stream>>>(XL, nullptr, nh_g, nh_b, nullptr, o_h, 0);  // h_new -> output

  // --- heads ---
  CVT(o_h, ABF0);
  GEMM(ABF0, WHP, XR, wh_b1, 256, 256, 0);
  ln_k<<<Bn / 8, 256, 0, stream>>>(XR, nullptr, wh_g, wh_bn, nullptr, T0, 1);  // wz
  head_w_k<<<gdiv(Bn, TPB), TPB, 0, stream>>>(T0, wh_W2, wh_b2, o_w);

  GEMM(ABF0, RHP, XR, rh_b1, 256, 256, 1);                                     // silu epilogue
  head_res_k<<<gdiv(Bn, TPB), TPB, 0, stream>>>(XR, rh_W2, rh_b2, o_res);

  Z(APACC, 2);
  GEMM(ABF0, APP, H128, ap_b1, 256, 128, 1);
  head_ap_k<<<gdiv(Bn, TPB), TPB, 0, stream>>>(H128, ap_W2, ap_b2, APACC);
  ap_final_k<<<1, 32, 0, stream>>>(APACC, o_ap);

  GEMM(ABF0, ADP, H128, ad_b1, 256, 128, 1);
  head_ad_k<<<gdiv(Bn, TPB), TPB, 0, stream>>>(H128, ad_W2, ad_b2, o_ad);
}
